// PrototypeNetwork_66769561584357
// MI455X (gfx1250) — matrix operand order repaired and checked by reading, not compiled, not run on hardware
//
#include <hip/hip_runtime.h>

// ---------------- problem constants ----------------
#define BN    4096      // batch rows
#define DD    1024      // feature dim
#define NCOLS 2800      // 400 + 800 + 1600 prototypes
#define G0    400
#define G1    1200      // group1 = [400,1200), group2 = [1200,2800)
#define NTILES (NCOLS / 16)   // 175 column tiles
#define HSPLIT 88             // half0 = tiles [0,88), half1 = [88,175)

typedef __attribute__((ext_vector_type(16))) __bf16 v16bf;
typedef __attribute__((ext_vector_type(8)))  float  v8f;

union Frag { uint4 q[2]; v16bf v; };

__device__ __forceinline__ unsigned int f2bf(float f) {
    // round-to-nearest-even fp32 -> bf16 (inputs are finite normals)
    unsigned int u = __float_as_uint(f);
    return (u + 0x7FFFu + ((u >> 16) & 1u)) >> 16;
}

// ---------------- prep: x -> bf16 ----------------
__global__ __launch_bounds__(256) void cvt_x_kernel(const float* __restrict__ x,
                                                    unsigned short* __restrict__ xb) {
    int i = (blockIdx.x * 256 + threadIdx.x) * 2;   // n = BN*DD, even
    unsigned int lo = f2bf(x[i]);
    unsigned int hi = f2bf(x[i + 1]);
    *reinterpret_cast<unsigned int*>(xb + i) = lo | (hi << 16);
}

// ---------------- prep: prototypes -> bf16 (concatenated) + fp32 row norms ----------------
__global__ __launch_bounds__(256) void proto_prep_kernel(const float* __restrict__ P0,
                                                         const float* __restrict__ P1,
                                                         const float* __restrict__ P2,
                                                         unsigned short* __restrict__ fb,
                                                         float* __restrict__ f2) {
    const int j = blockIdx.x;                        // 0..2799
    const float* src;
    if (j < G0)       src = P0 + (size_t)j * DD;
    else if (j < G1)  src = P1 + (size_t)(j - G0) * DD;
    else              src = P2 + (size_t)(j - G1) * DD;

    __shared__ float red[256];
    const int t = threadIdx.x;
    float4 v = reinterpret_cast<const float4*>(src)[t];            // 256*4 = 1024 floats
    float acc = v.x * v.x + v.y * v.y + v.z * v.z + v.w * v.w;
    unsigned int p0 = f2bf(v.x) | (f2bf(v.y) << 16);
    unsigned int p1 = f2bf(v.z) | (f2bf(v.w) << 16);
    uint2 pk; pk.x = p0; pk.y = p1;
    reinterpret_cast<uint2*>(fb + (size_t)j * DD)[t] = pk;
    red[t] = acc;
    __syncthreads();
    for (int s = 128; s > 0; s >>= 1) {
        if (t < s) red[t] += red[t + s];
        __syncthreads();
    }
    if (t == 0) f2[j] = red[0];
}

// ---------------- fused WMMA GEMM + per-group argmin ----------------
// grid = (256 M tiles, 2 column halves), block = 256 (8 waves of 32)
__global__ __launch_bounds__(256) void wmma_argmin_kernel(const unsigned short* __restrict__ xb,
                                                          const unsigned short* __restrict__ fb,
                                                          const float* __restrict__ f2,
                                                          float* __restrict__ hv,
                                                          int* __restrict__ hi) {
    __shared__ float s_v[3][8][16][16];
    __shared__ int   s_i[3][8][16][16];

    const int wave = threadIdx.x >> 5;
    const int lane = threadIdx.x & 31;
    const int l15  = lane & 15;
    const int m0   = blockIdx.x * 16;
    const int half = blockIdx.y;
    const int hiA  = (lane < 16) ? 0 : 2;   // A frag uint4 selector (contiguous k = 16h + s, matches B)
    const int hiB  = (lane < 16) ? 0 : 2;   // B frag uint4 selector (contiguous k = 16h + s, matches A)
    const size_t abase = (size_t)(m0 + l15) * DD;

    // --- A fragments: load once, keep resident (32 x 8 VGPRs) ---
    Frag afrag[32];
#pragma unroll
    for (int ks = 0; ks < 32; ++ks) {
        const uint4* ap = reinterpret_cast<const uint4*>(xb + abase + ks * 32);
        afrag[ks].q[0] = ap[hiA];
        afrag[ks].q[1] = ap[hiA + 1];
    }

    float bestv[3][8];
    int   besti[3][8];
#pragma unroll
    for (int g = 0; g < 3; ++g)
#pragma unroll
        for (int r = 0; r < 8; ++r) { bestv[g][r] = 3.0e38f; besti[g][r] = 0x7FFFFFFF; }

    const int jtBeg = (half == 0) ? 0 : HSPLIT;
    const int jtEnd = (half == 0) ? HSPLIT : NTILES;

    for (int jt = jtBeg + wave; jt < jtEnd; jt += 8) {   // wave-uniform control flow
        const int j0  = jt * 16;
        const int col = j0 + l15;
        const unsigned short* bptr = fb + (size_t)col * DD;

        // --- software pipeline: 4 rotating B buffers, prefetch distance 3 ---
        Frag bq[4];
#pragma unroll
        for (int p = 0; p < 3; ++p) {
            const uint4* bp = reinterpret_cast<const uint4*>(bptr + p * 32);
            bq[p].q[0] = bp[hiB];
            bq[p].q[1] = bp[hiB + 1];
        }
        v8f c = {0.f, 0.f, 0.f, 0.f, 0.f, 0.f, 0.f, 0.f};
#pragma unroll
        for (int ks = 0; ks < 32; ++ks) {
            if (ks + 3 < 32) {   // issue load 3 steps ahead BEFORE this step's WMMA
                const uint4* bp = reinterpret_cast<const uint4*>(bptr + (ks + 3) * 32);
                bq[(ks + 3) & 3].q[0] = bp[hiB];
                bq[(ks + 3) & 3].q[1] = bp[hiB + 1];
            }
            c = __builtin_amdgcn_wmma_f32_16x16x32_bf16(false, afrag[ks].v, false, bq[ks & 3].v,
                                                        (short)0, c, false, false);
        }

        const float fn = f2[col];
        const int g      = (j0 < G0) ? 0 : ((j0 < G1) ? 1 : 2);
        const int gstart = (g == 0) ? 0 : ((g == 1) ? G0 : G1);
        const int rel    = col - gstart;
#pragma unroll
        for (int r = 0; r < 8; ++r) {
            float val = fn - 2.0f * c[r];
            if (val < bestv[g][r] || (val == bestv[g][r] && rel < besti[g][r])) {
                bestv[g][r] = val;
                besti[g][r] = rel;
            }
        }
    }

    const int rbase = (lane < 16) ? 0 : 8;   // C/D layout: lanes 0-15 -> M=r, lanes 16-31 -> M=8+r
#pragma unroll
    for (int g = 0; g < 3; ++g)
#pragma unroll
        for (int r = 0; r < 8; ++r) {
            s_v[g][wave][rbase + r][l15] = bestv[g][r];
            s_i[g][wave][rbase + r][l15] = besti[g][r];
        }
    __syncthreads();

    if (threadIdx.x < 48) {
        const int g   = threadIdx.x >> 4;
        const int row = threadIdx.x & 15;
        float bv = 3.0e38f; int bi = 0x7FFFFFFF;
        for (int w = 0; w < 8; ++w)
            for (int cc = 0; cc < 16; ++cc) {
                float v = s_v[g][w][row][cc];
                int   i = s_i[g][w][row][cc];
                if (v < bv || (v == bv && i < bi)) { bv = v; bi = i; }
            }
        hv[(half * 3 + g) * BN + (m0 + row)] = bv;
        hi[(half * 3 + g) * BN + (m0 + row)] = bi;
    }
}

// ---------------- gather + write outputs (HBM-write bound) ----------------
// grid = 4096 (one block per batch row), block = 256
__global__ __launch_bounds__(256) void gather_kernel(const float* __restrict__ P0,
                                                     const float* __restrict__ P1,
                                                     const float* __restrict__ P2,
                                                     const float* __restrict__ hv,
                                                     const int* __restrict__ hi,
                                                     float* __restrict__ out) {
    const int b = blockIdx.x;
    const int t = threadIdx.x;

    // merge the two column-half argmin results per group
    int idx[3];
#pragma unroll
    for (int g = 0; g < 3; ++g) {
        float v0 = hv[g * BN + b];
        float v1 = hv[(3 + g) * BN + b];
        int   i0 = hi[g * BN + b];
        int   i1 = hi[(3 + g) * BN + b];
        idx[g] = (v1 < v0 || (v1 == v0 && i1 < i0)) ? i1 : i0;
    }
    const int i0 = idx[0], i1 = idx[1], i2 = idx[2];
    const int c0 = i0 >> 2;    // K=4
    const int c1 = i1 >> 3;    // K=8
    const int c2 = i2 >> 4;    // K=16

    const size_t Q = DD / 4;   // 256 float4 per row; one per thread
    const float4* p0 = reinterpret_cast<const float4*>(P0);
    const float4* p1 = reinterpret_cast<const float4*>(P1);
    const float4* p2 = reinterpret_cast<const float4*>(P2);
    float4* o = reinterpret_cast<float4*>(out);

    // nearest_prototypes [B,3,D]
    size_t dst = (size_t)b * 3 * Q;
    o[dst + 0 * Q + t] = p0[(size_t)i0 * Q + t];
    o[dst + 1 * Q + t] = p1[(size_t)i1 * Q + t];
    o[dst + 2 * Q + t] = p2[(size_t)i2 * Q + t];

    // prototype_set [B,28,D]
    size_t sb = (size_t)BN * 3 * Q + (size_t)b * 28 * Q;
#pragma unroll
    for (int k = 0; k < 4; ++k)
        o[sb + (size_t)k * Q + t] = p0[(size_t)(c0 * 4 + k) * Q + t];
#pragma unroll
    for (int k = 0; k < 8; ++k)
        o[sb + (size_t)(4 + k) * Q + t] = p1[(size_t)(c1 * 8 + k) * Q + t];
#pragma unroll
    for (int k = 0; k < 16; ++k)
        o[sb + (size_t)(12 + k) * Q + t] = p2[(size_t)(c2 * 16 + k) * Q + t];

    // class indices from last group, as float
    if (t == 0) out[(size_t)BN * 31 * DD + b] = (float)c2;
}

// ---------------- host-side launcher ----------------
extern "C" void kernel_launch(void* const* d_in, const int* in_sizes, int n_in,
                              void* d_out, int out_size, void* d_ws, size_t ws_size,
                              hipStream_t stream) {
    (void)in_sizes; (void)n_in; (void)out_size; (void)ws_size;
    const float* x  = (const float*)d_in[0];
    const float* P0 = (const float*)d_in[1];
    const float* P1 = (const float*)d_in[2];
    const float* P2 = (const float*)d_in[3];
    float* out = (float*)d_out;

    // workspace layout (16-byte aligned slices)
    unsigned short* xb = (unsigned short*)d_ws;                        // BN*DD bf16  = 8 MiB
    unsigned short* fb = xb + (size_t)BN * DD;                         // NCOLS*DD bf16
    float*          f2 = (float*)(fb + (size_t)NCOLS * DD);            // NCOLS fp32
    float*          hv = f2 + NCOLS;                                   // [2][3][BN] fp32
    int*            hi = (int*)(hv + 2 * 3 * BN);                      // [2][3][BN] int32

    cvt_x_kernel<<<(BN * DD) / 512, 256, 0, stream>>>(x, xb);
    proto_prep_kernel<<<NCOLS, 256, 0, stream>>>(P0, P1, P2, fb, f2);
    dim3 grid(BN / 16, 2);
    wmma_argmin_kernel<<<grid, 256, 0, stream>>>(xb, fb, f2, hv, hi);
    gather_kernel<<<BN, 256, 0, stream>>>(P0, P1, P2, hv, hi, out);
}
